// DynamicGraphAttension_8229157339662
// MI455X (gfx1250) — compile-verified, matrix operand order checked by reading, not run on hardware
//
#include <hip/hip_runtime.h>

typedef __attribute__((ext_vector_type(16))) __bf16 v16bf;
typedef __attribute__((ext_vector_type(8)))  __bf16 v8bf;
typedef __attribute__((ext_vector_type(4)))  __bf16 v4bf;
typedef __attribute__((ext_vector_type(8)))  float  v8f;

#define NB      32
#define SEQ     4096
#define KD      1024
#define VD      1024
#define LFP     512
#define NCHUNK  8
#define SCHUNK  512            // rows of S per block
#define RM      64             // rows staged per supertile (4 M-tiles)
#define MT      4              // M-tiles per supertile
#define NT      4              // N-tiles per wave (8 waves * 4 * 16 = 512 = LFP)
#define NSUPER  (SCHUNK / RM)  // 8
#define LDS_STRIDE (VD + 8)    // +16B pad: kills bank conflicts on fragment loads

union FragU { v16bf v; v8bf h[2]; };

__device__ __forceinline__ float fast_tanh(float x) {
#if defined(__HIP_DEVICE_COMPILE__)
# if __has_builtin(__builtin_amdgcn_tanhf)
  return __builtin_amdgcn_tanhf(x);          // V_TANH_F32
# else
  float t = __expf(2.0f * x);                // V_EXP_F32
  return 1.0f - 2.0f * __builtin_amdgcn_rcpf(t + 1.0f);
# endif
#else
  return tanhf(x);
#endif
}

// ---------------- key projection: key_p[b][l] = key[b,:] . W_key[:,l] ------
__global__ void keyp_kernel(const float* __restrict__ key,
                            const float* __restrict__ Wk,
                            float* __restrict__ keyp) {
  int b = blockIdx.x;
  int l = threadIdx.x;                 // 512 threads
  const float* kr = key + (size_t)b * KD;
  float acc = 0.f;
  for (int k = 0; k < KD; ++k)
    acc = fmaf(kr[k], Wk[(size_t)k * LFP + l], acc);
  keyp[(size_t)b * LFP + l] = acc;
}

// ---- W_value transpose + bf16: wt[l][v] = (bf16) W_value[v][l] -----------
__global__ void wt_kernel(const float* __restrict__ Wv, __bf16* __restrict__ wt) {
  int idx = blockIdx.x * 256 + threadIdx.x;     // LFP*VD total
  int l = idx >> 10;                            // /VD
  int v = idx & (VD - 1);
  wt[idx] = (__bf16)Wv[(size_t)v * LFP + l];
}

// ---------------- fused: WMMA projection + tanh scores + online softmax ----
__global__ __launch_bounds__(256)
void fused_attn_kernel(const float* __restrict__ value,
                       const __bf16* __restrict__ wt,
                       const float* __restrict__ keyp,
                       const float* __restrict__ wattn,
                       float* __restrict__ scores_g,
                       float* __restrict__ pctx,
                       float* __restrict__ pml) {
  __shared__ __bf16 vtile[RM * LDS_STRIDE];     // 132 KB staged value rows (bf16)
  __shared__ float  c_loc[VD];                  // running context accumulator
  __shared__ float  score_sh[RM];
  __shared__ float  p_sh[RM];
  __shared__ float  mrun_sh, lrun_sh, scale_sh;

  const int chunk = blockIdx.x;                 // 0..NCHUNK-1
  const int b     = blockIdx.y;                 // 0..NB-1
  const int tid   = threadIdx.x;
  const int wid   = tid >> 5;                   // 8 waves
  const int lane  = tid & 31;
  const int lr    = lane & 15;
  const int half  = lane >> 4;

  // loop-invariant per-lane epilogue scalars: n = wid*64 + nt*16 + lr
  float kp_r[NT], wa_r[NT];
#pragma unroll
  for (int nt = 0; nt < NT; ++nt) {
    const int n = wid * 64 + nt * 16 + lr;
    kp_r[nt] = keyp[(size_t)b * LFP + n];
    wa_r[nt] = wattn[n];
  }

  // init accumulators
  {
    int v = tid * 4;
    c_loc[v] = 0.f; c_loc[v+1] = 0.f; c_loc[v+2] = 0.f; c_loc[v+3] = 0.f;
  }
  if (tid == 0) { mrun_sh = -3.0e38f; lrun_sh = 0.f; }
  __syncthreads();

  for (int st = 0; st < NSUPER; ++st) {
    const int s0 = chunk * SCHUNK + st * RM;

    // ---- stage 64 x 1024 value rows into LDS, fp32 -> bf16 ----
    const float* src = value + ((size_t)b * SEQ + s0) * VD;
    for (int i = tid; i < (RM * VD) / 4; i += 256) {
      int e = i * 4;
      int r = e >> 10;
      int c = e & (VD - 1);
      float4 f = *(const float4*)(src + e);
      v4bf pk = { (__bf16)f.x, (__bf16)f.y, (__bf16)f.z, (__bf16)f.w };
      *(v4bf*)&vtile[r * LDS_STRIDE + c] = pk;   // ds_store_b64
    }
    if (tid < RM) score_sh[tid] = 0.f;
    __syncthreads();

    // ---- WMMA: val_p block [64 x 512]; wave = 4Mx4N register tile ----
    {
      v8f acc[MT][NT];
#pragma unroll
      for (int mt = 0; mt < MT; ++mt)
#pragma unroll
        for (int nt = 0; nt < NT; ++nt)
          acc[mt][nt] = (v8f){0.f,0.f,0.f,0.f,0.f,0.f,0.f,0.f};

      const __bf16* wrow0 = wt + (size_t)(wid * 64 + lr) * VD;

#pragma unroll 2
      for (int k = 0; k < VD; k += 32) {
        const int ko = k + half * 8;
        FragU a[MT], bm[NT];
#pragma unroll
        for (int mt = 0; mt < MT; ++mt) {
          const __bf16* ar = &vtile[(mt * 16 + lr) * LDS_STRIDE + ko];
          a[mt].h[0] = *(const v8bf*)(ar);
          a[mt].h[1] = *(const v8bf*)(ar + 16);
        }
#pragma unroll
        for (int nt = 0; nt < NT; ++nt) {
          const __bf16* br = wrow0 + (size_t)nt * 16 * VD + ko;
          bm[nt].h[0] = *(const v8bf*)(br);
          bm[nt].h[1] = *(const v8bf*)(br + 16);
        }
#pragma unroll
        for (int mt = 0; mt < MT; ++mt)
#pragma unroll
          for (int nt = 0; nt < NT; ++nt)
            acc[mt][nt] = __builtin_amdgcn_wmma_f32_16x16x32_bf16(
                false, a[mt].v, false, bm[nt].v, (short)0, acc[mt][nt],
                false, false);
      }

      // epilogue: tanh(val_p + key_p) * W_attn; fold NT in registers, then
      // one 16-lane shuffle reduction + one LDS atomic per (mt, r)
#pragma unroll
      for (int mt = 0; mt < MT; ++mt) {
#pragma unroll
        for (int r = 0; r < 8; ++r) {
          float x = 0.f;
#pragma unroll
          for (int nt = 0; nt < NT; ++nt)
            x = fmaf(fast_tanh(acc[mt][nt][r] + kp_r[nt]), wa_r[nt], x);
          x += __shfl_xor(x, 1); x += __shfl_xor(x, 2);
          x += __shfl_xor(x, 4); x += __shfl_xor(x, 8);
          if (lr == 0) atomicAdd(&score_sh[mt * 16 + r + half * 8], x);
        }
      }
    }
    __syncthreads();

    // ---- raw scores out + online softmax stats (wave 0, parallel) ----
    if (tid < RM)
      scores_g[(size_t)b * SEQ + s0 + tid] = score_sh[tid];
    if (wid == 0) {
      const float mprev = mrun_sh;
      float sA = score_sh[lane];
      float sB = score_sh[lane + 32];
      float mnew = fmaxf(sA, sB);
#pragma unroll
      for (int off = 16; off > 0; off >>= 1)
        mnew = fmaxf(mnew, __shfl_xor(mnew, off));
      mnew = fmaxf(mnew, mprev);
      float pA = __expf(sA - mnew);
      float pB = __expf(sB - mnew);
      p_sh[lane]      = pA;
      p_sh[lane + 32] = pB;
      float ls = pA + pB;
#pragma unroll
      for (int off = 16; off > 0; off >>= 1)
        ls += __shfl_xor(ls, off);
      if (lane == 0) {
        const float sc = __expf(mprev - mnew);
        lrun_sh  = lrun_sh * sc + ls;
        mrun_sh  = mnew;
        scale_sh = sc;
      }
    }
    __syncthreads();

    // ---- fold staged rows into running context (value never re-read) ----
    {
      const float sc = scale_sh;
      const int v = tid * 4;
      float c0 = c_loc[v] * sc, c1 = c_loc[v+1] * sc,
            c2 = c_loc[v+2] * sc, c3 = c_loc[v+3] * sc;
#pragma unroll 8
      for (int r = 0; r < RM; ++r) {
        const float pr = p_sh[r];
        v4bf vr = *(const v4bf*)&vtile[r * LDS_STRIDE + v];
        c0 = fmaf(pr, (float)vr[0], c0);
        c1 = fmaf(pr, (float)vr[1], c1);
        c2 = fmaf(pr, (float)vr[2], c2);
        c3 = fmaf(pr, (float)vr[3], c3);
      }
      c_loc[v] = c0; c_loc[v+1] = c1; c_loc[v+2] = c2; c_loc[v+3] = c3;
    }
    __syncthreads();   // before next supertile overwrites vtile
  }

  // ---- write per-chunk partials ----
  const int pi = b * NCHUNK + chunk;
  for (int i = tid; i < VD; i += 256)
    pctx[(size_t)pi * VD + i] = c_loc[i];
  if (tid == 0) {
    pml[pi * 2]     = mrun_sh;
    pml[pi * 2 + 1] = lrun_sh;
  }
}

// ---------------- combine partials -> context + normalized weights --------
__global__ void combine_kernel(const float* __restrict__ pctx,
                               const float* __restrict__ pml,
                               const float* __restrict__ scores_g,
                               float* __restrict__ out) {
  const int b = blockIdx.x;
  const int tid = threadIdx.x;          // 256
  float M = -3.0e38f;
  for (int i = 0; i < NCHUNK; ++i) M = fmaxf(M, pml[(b * NCHUNK + i) * 2]);
  float L = 0.f;
  float ef[NCHUNK];
  for (int i = 0; i < NCHUNK; ++i) {
    ef[i] = __expf(pml[(b * NCHUNK + i) * 2] - M);
    L += pml[(b * NCHUNK + i) * 2 + 1] * ef[i];
  }
  const float invL = 1.0f / L;

  for (int v = tid; v < VD; v += 256) {
    float s = 0.f;
    for (int i = 0; i < NCHUNK; ++i)
      s = fmaf(ef[i], pctx[(size_t)(b * NCHUNK + i) * VD + v], s);
    out[(size_t)b * VD + v] = s * invL;
  }
  float* wout = out + (size_t)NB * VD;
  for (int s = tid; s < SEQ; s += 256)
    wout[(size_t)b * SEQ + s] = __expf(scores_g[(size_t)b * SEQ + s] - M) * invL;
}

extern "C" void kernel_launch(void* const* d_in, const int* in_sizes, int n_in,
                              void* d_out, int out_size, void* d_ws, size_t ws_size,
                              hipStream_t stream) {
  const float* key   = (const float*)d_in[0];   // [32,1024]
  const float* value = (const float*)d_in[1];   // [32,4096,1024]
  const float* Wk    = (const float*)d_in[2];   // [1024,512]
  const float* Wv    = (const float*)d_in[3];   // [1024,512]
  const float* Wa    = (const float*)d_in[4];   // [512]
  float* out = (float*)d_out;

  char* ws = (char*)d_ws;
  float*  keyp   = (float*)ws;                                 //  64 KB
  __bf16* wt     = (__bf16*)(ws + 65536);                      //   1 MB
  float*  scores = (float*)(ws + 65536 + 1048576);             // 512 KB
  float*  pctx   = (float*)(ws + 65536 + 1048576 + 524288);    //   1 MB
  float*  pml    = (float*)(ws + 65536 + 1048576 + 524288 + 1048576);

  keyp_kernel<<<NB, LFP, 0, stream>>>(key, Wk, keyp);
  wt_kernel<<<(LFP * VD) / 256, 256, 0, stream>>>(Wv, wt);

  dim3 grid(NCHUNK, NB);
  fused_attn_kernel<<<grid, 256, 0, stream>>>(value, wt, keyp, Wa,
                                              scores, pctx, pml);
  combine_kernel<<<NB, 256, 0, stream>>>(pctx, pml, scores, out);
}
